// StackEncoder_7456063226108
// MI455X (gfx1250) — matrix operand order repaired and checked by reading, not compiled, not run on hardware
//
#include <hip/hip_runtime.h>
#include <stdint.h>
#include <math.h>

// ---------------------------------------------------------------------------
// SPINN stack-encoder for MI455X (gfx1250, wave32, WMMA).
//
//  * Precompute shift/reduce stack-pointer schedule (tiny sequential kernel).
//  * Pack all weights once into WMMA B-fragment lane order (bf16).
//  * Persistent kernel: 16 WGs x 256 threads (8 waves). WG w owns tracking
//    column-tile w and the 4 tree-gate tiles for column block w, so all four
//    gate accumulators for a (row,col) live in one wave's registers.
//  * Per-WG weight slice (98KB) staged into LDS once; inner loop reads
//    B-fragments with ds_read_b128, A-fragments with per-lane global b128
//    gathers from bf16 activation stacks.
//  * v_wmma_f32_16x16x32_bf16, f32 accumulate. All WMMA chains start from
//    inline-zero C (biases folded into the elementwise stages) to avoid
//    accumulator splat/copy trains. Stage-A slab loops are unroll-capped to
//    keep the wave under the 256-VGPR direct-addressing boundary.
//  * 3 grid barriers per step (sense-reversing, device-scope atomics).
// ---------------------------------------------------------------------------

typedef __attribute__((ext_vector_type(16))) __bf16 v16bf;
typedef __attribute__((ext_vector_type(8)))  float  v8f;
typedef __bf16 bf16_t;

constexpr int kB   = 128;
constexpr int kL   = 256;
constexpr int kD   = 256;
constexpr int kTD  = 64;
constexpr int kT   = 2 * kL - 1;   // 511 transitions
constexpr int kTP1 = kT + 1;       // 512 stack rows (row 0 = zeros)

constexpr int NWG      = 16;
constexpr int NTHREADS = 256;      // 8 wave32 waves

// K-slabs of 32 per column tile
constexpr int SLABS_TRK   = 26;    // 768 (h_b,h_r,h_l) + 64 (th)
constexpr int SLABS_TREE  = 18;    // 256 (h_l) + 256 (h_r) + 64 (x)
constexpr int SLABS_PER_WG = SLABS_TRK + 4 * SLABS_TREE;          // 98
constexpr int TOTAL_SLABS  = 16 * SLABS_TRK + 64 * SLABS_TREE;    // 1568
constexpr uint32_t LDS_BYTES = (uint32_t)SLABS_PER_WG * 32 * 32;  // 100352 B

struct ParamsPtrs {
  const float *W_i, *W_o, *W_u;
  const float *U_il, *U_ir, *U_fl, *U_fr, *U_ol, *U_or, *U_ul, *U_ur;
  const float *b_i, *b_f, *b_o, *b_u;
  const float *W_ih, *W_hh, *b_ih, *b_hh;
};

struct WsPtrs {
  bf16_t*  stk_h;   // [B][T+1][D]  bf16 stack of hidden states (WMMA A source)
  float*   stk_c;   // [B][T+1][D]  f32 stack of cell states
  bf16_t*  tok_h;   // [B][L][D]    bf16 tokens_h
  bf16_t*  packW;   // [TOTAL_SLABS][32 lanes][16] packed B-fragments
  bf16_t*  x_bf;    // [B][TD]      tracking hidden (th), bf16
  float*   tc;      // [B][TD]      tracking cell, f32
  float*   Gbuf;    // [B][256]     tracking pre-activation gates (no bias)
  int *sp1, *sp2, *bp, *msk;   // [T+1][B] schedules
  int *ptrq;                   // [B][T+1] scratch for schedule kernel
  unsigned* bar;               // {count, gen}
};

// ---------------------------------------------------------------------------

__device__ __forceinline__ float sigmoidf(float x) {
  return 1.0f / (1.0f + __expf(-x));
}

__device__ __forceinline__ v16bf frag_cast(uint4 lo, uint4 hi) {
  union { uint4 u[2]; v16bf v; } c;
  c.u[0] = lo; c.u[1] = hi;
  return c.v;
}

// A-fragment (16x32 bf16, M x K): this lane holds row m, K values
// {k0..k0+7} and {k0+16..k0+23} where k0 = slab*32 + (lane>=16 ? 8 : 0).
__device__ __forceinline__ v16bf load_a(const bf16_t* row, int k0) {
  const uint4* p0 = reinterpret_cast<const uint4*>(row + k0);
  const uint4* p1 = reinterpret_cast<const uint4*>(row + k0 + 16);
  return frag_cast(*p0, *p1);
}

// B-fragment (32x16 bf16, K x N) pre-packed per lane: 16 contiguous bf16.
__device__ __forceinline__ v16bf load_b(const uint4* smem, int ls, int lane) {
  const uint4* p = smem + ((size_t)ls * 32 + lane) * 2;
  return frag_cast(p[0], p[1]);
}

__device__ __forceinline__ v8f wmma_bf16(v16bf a, v16bf b, v8f c) {
  return __builtin_amdgcn_wmma_f32_16x16x32_bf16(
      /*neg_a=*/false, a, /*neg_b=*/false, b,
      /*c_mod=*/(short)0, c, /*reuse_a=*/false, /*reuse_b=*/false);
}

// First WMMA of a chain: C = inline zero (no register tuple to materialize).
__device__ __forceinline__ v8f wmma_bf16_z(v16bf a, v16bf b) {
  v8f z = {0.f, 0.f, 0.f, 0.f, 0.f, 0.f, 0.f, 0.f};
  return __builtin_amdgcn_wmma_f32_16x16x32_bf16(
      false, a, false, b, (short)0, z, false, false);
}

__device__ __forceinline__ void grid_barrier(unsigned* count, unsigned* gen,
                                             unsigned nwg) {
  __threadfence();
  __syncthreads();
  if (threadIdx.x == 0) {
    unsigned g = __hip_atomic_load(gen, __ATOMIC_ACQUIRE, __HIP_MEMORY_SCOPE_AGENT);
    unsigned c = __hip_atomic_fetch_add(count, 1u, __ATOMIC_ACQ_REL,
                                        __HIP_MEMORY_SCOPE_AGENT);
    if (c + 1 == nwg) {
      __hip_atomic_store(count, 0u, __ATOMIC_RELAXED, __HIP_MEMORY_SCOPE_AGENT);
      __hip_atomic_fetch_add(gen, 1u, __ATOMIC_RELEASE, __HIP_MEMORY_SCOPE_AGENT);
    } else {
      while (__hip_atomic_load(gen, __ATOMIC_ACQUIRE,
                               __HIP_MEMORY_SCOPE_AGENT) == g) {
        __builtin_amdgcn_s_sleep(1);
      }
    }
  }
  __syncthreads();
}

// ---------------------------------------------------------------------------
// Schedule kernel: per-batch sequential walk of the shift/reduce transitions,
// reproducing the reference's pointer-queue logic exactly.
__global__ void sched_kernel(const int* __restrict__ trans, int* sp1, int* sp2,
                             int* bp, int* msk, int* ptrq) {
  int b = threadIdx.x;
  if (b >= kB) return;
  int* q = ptrq + (size_t)b * kTP1;
  for (int i = 0; i < kTP1; ++i) q[i] = 0;
  int depth = 0, bufp = 0;
  for (int t = 1; t <= kT; ++t) {
    int m  = trans[(size_t)b * kT + (t - 1)];
    int s1 = (depth >= 1) ? q[depth - 1] : 0;
    int s2 = (depth >= 2) ? q[depth - 2] : 0;
    sp1[t * kB + b] = s1;
    sp2[t * kB + b] = s2;
    int bpp = bufp < 0 ? 0 : (bufp > kL - 1 ? kL - 1 : bufp);
    bp[t * kB + b]  = bpp;
    msk[t * kB + b] = m;
    bufp += (1 - m);
    int d2 = depth - 2 * m;
    int ci = d2 < 0 ? 0 : (d2 > kT ? kT : d2);
    q[ci] = t;
    depth = d2 + 1;
  }
}

// ---------------------------------------------------------------------------
// Weight packer: one block per column tile.
//   Tiles 0..15  : tracking gates, N = tile*16.., K = [h_b|h_r|h_l](W_ih) + th(W_hh)
//   Tiles 16..79 : tree gate g=(tile-16)/16 in {i,o,f,u}, col block (tile-16)%16,
//                  K = h_l(U_gl) + h_r(U_gr) + x(W_g).  f uses W_o (ref bug kept).
// B-fragment lane order: lane n = lane&15, K half = lane>>4 (16 values each).
__global__ void pack_weights(ParamsPtrs P, bf16_t* __restrict__ packW) {
  int tile = blockIdx.x;
  int lane = threadIdx.x & 31;
  int nslab = (tile < 16) ? SLABS_TRK : SLABS_TREE;
  int nbase = (tile < 16) ? tile * 16 : ((tile - 16) & 15) * 16;
  int n = nbase + (lane & 15);
  const float* Ul[4] = {P.U_il, P.U_ol, P.U_fl, P.U_ul};
  const float* Ur[4] = {P.U_ir, P.U_or, P.U_fr, P.U_ur};
  const float* Wx[4] = {P.W_i,  P.W_o,  P.W_o,  P.W_u};  // f-gate uses W_o
  int g = (tile >= 16) ? ((tile - 16) >> 4) : 0;

  for (int slab = threadIdx.x >> 5; slab < nslab; slab += (blockDim.x >> 5)) {
    size_t gslab = (tile < 16)
        ? (size_t)tile * SLABS_TRK + slab
        : (size_t)16 * SLABS_TRK + (size_t)(tile - 16) * SLABS_TREE + slab;
    bf16_t* dst = packW + (gslab * 32 + lane) * 16;
    for (int j = 0; j < 16; ++j) {
      int kk = ((lane >> 4) * 8) + (j & 7) + ((j >> 3) * 16);      // 0..31 within slab
      int k  = slab * 32 + kk;
      float v;
      if (tile < 16) {
        v = (k < 768) ? P.W_ih[(size_t)n * 768 + k]
                      : P.W_hh[(size_t)n * 64 + (k - 768)];
      } else if (slab < 8) {
        v = Ul[g][(size_t)n * 256 + k];
      } else if (slab < 16) {
        v = Ur[g][(size_t)n * 256 + (k - 256)];
      } else {
        v = Wx[g][(size_t)n * 64 + (k - 512)];
      }
      dst[j] = (bf16_t)v;
    }
  }
}

// ---------------------------------------------------------------------------
// Init: tokens_h -> bf16, zero stack row 0, seed tracking state, reset barrier.
__global__ void init_kernel(const float* __restrict__ tokens_h,
                            const float* __restrict__ th0,
                            const float* __restrict__ tc0,
                            bf16_t* tok_h, bf16_t* stk_h, float* stk_c,
                            bf16_t* x_bf, float* tc, unsigned* bar) {
  size_t i = (size_t)blockIdx.x * blockDim.x + threadIdx.x;
  if (i < (size_t)kB * kL * kD) tok_h[i] = (bf16_t)tokens_h[i];
  if (i < (size_t)kB * kD) {
    size_t b = i / kD, d = i % kD;
    stk_h[(b * kTP1) * kD + d] = (bf16_t)0.0f;
    stk_c[(b * kTP1) * kD + d] = 0.0f;
  }
  if (i < (size_t)kB * kTD) {
    x_bf[i] = (bf16_t)th0[i];
    tc[i]   = tc0[i];
  }
  if (i == 0) { bar[0] = 0u; bar[1] = 0u; }
}

// ---------------------------------------------------------------------------
// Persistent main kernel.
__global__ __launch_bounds__(NTHREADS, 1)
void spinn_persistent(ParamsPtrs P, WsPtrs W,
                      const float* __restrict__ tokens_h_f32,
                      const float* __restrict__ tokens_c_f32,
                      float* __restrict__ out) {
  extern __shared__ uint4 smem[];
  const int wg   = blockIdx.x;           // 0..15
  const int lane = threadIdx.x & 31;
  const int wave = threadIdx.x >> 5;     // 0..7, owns M-tile = wave

  // ---- Stage-in this WG's constant weight slice into LDS (once) ----
  for (int u = threadIdx.x; u < SLABS_PER_WG * 32; u += NTHREADS) {
    int ls = u >> 5, ln = u & 31;
    size_t gslab;
    if (ls < SLABS_TRK) {
      gslab = (size_t)wg * SLABS_TRK + ls;
    } else {
      int r = ls - SLABS_TRK;
      int g = r / SLABS_TREE, s = r % SLABS_TREE;
      gslab = (size_t)16 * SLABS_TRK + (size_t)(g * 16 + wg) * SLABS_TREE + s;
    }
    const uint4* src = reinterpret_cast<const uint4*>(W.packW) + (gslab * 32 + ln) * 2;
    uint4* dst = smem + ((size_t)ls * 32 + ln) * 2;
    dst[0] = src[0];
    dst[1] = src[1];
  }
  __syncthreads();

  const int colLo = lane & 15;           // N within 16-wide tile
  const int khalf = lane >> 4;           // K-half select for A fragments
  const int aRow  = wave * 16 + colLo;   // batch row this lane gathers A data for
  const int col   = wg * 16 + colLo;     // output column (0..255) in C fragments
  const int k0a   = khalf * 8;           // A k-offset inside a 32-slab

  // Tree biases applied in stage C (not in the accumulators).
  const float bT0 = P.b_i[col], bT1 = P.b_o[col], bT2 = P.b_f[col], bT3 = P.b_u[col];

  unsigned* barCount = W.bar;
  unsigned* barGen   = W.bar + 1;
  const bf16_t* xrow = W.x_bf + (size_t)aRow * kTD;

  for (int t = 1; t <= kT; ++t) {
    const int s1 = W.sp1[t * kB + aRow];
    const int s2 = W.sp2[t * kB + aRow];
    const int bq = W.bp [t * kB + aRow];
    const bf16_t* rowHb = W.tok_h + ((size_t)aRow * kL   + bq) * kD;
    const bf16_t* rowHr = W.stk_h + ((size_t)aRow * kTP1 + s1) * kD;
    const bf16_t* rowHl = W.stk_h + ((size_t)aRow * kTP1 + s2) * kD;
    if (t < kT) __builtin_prefetch(&W.sp1[(t + 1) * kB + aRow], 0, 0);

    // ---- Stage A: tracking gates, K = 832 = [h_b|h_r|h_l|th] ----
    v8f accG = wmma_bf16_z(load_a(rowHb, k0a), load_b(smem, 0, lane));
#pragma unroll 2
    for (int s = 1; s < 8; ++s)
      accG = wmma_bf16(load_a(rowHb, s * 32 + k0a), load_b(smem, s, lane), accG);
#pragma unroll 2
    for (int s = 0; s < 8; ++s)
      accG = wmma_bf16(load_a(rowHr, s * 32 + k0a), load_b(smem, 8 + s, lane), accG);
#pragma unroll 2
    for (int s = 0; s < 8; ++s)
      accG = wmma_bf16(load_a(rowHl, s * 32 + k0a), load_b(smem, 16 + s, lane), accG);
#pragma unroll
    for (int s = 0; s < 2; ++s)
      accG = wmma_bf16(load_a(xrow, s * 32 + k0a), load_b(smem, 24 + s, lane), accG);

    // ---- Stage A: tree gates, h_l / h_r contributions (A reused 4x) ----
    v8f accT[4];
    {
      v16bf a0 = load_a(rowHl, k0a);
#pragma unroll
      for (int g = 0; g < 4; ++g)
        accT[g] = wmma_bf16_z(a0, load_b(smem, SLABS_TRK + g * SLABS_TREE, lane));
    }
#pragma unroll 2
    for (int s = 1; s < 8; ++s) {
      v16bf a = load_a(rowHl, s * 32 + k0a);
#pragma unroll
      for (int g = 0; g < 4; ++g)
        accT[g] = wmma_bf16(a, load_b(smem, SLABS_TRK + g * SLABS_TREE + s, lane), accT[g]);
    }
#pragma unroll 2
    for (int s = 0; s < 8; ++s) {
      v16bf a = load_a(rowHr, s * 32 + k0a);
#pragma unroll
      for (int g = 0; g < 4; ++g)
        accT[g] = wmma_bf16(a, load_b(smem, SLABS_TRK + g * SLABS_TREE + 8 + s, lane), accT[g]);
    }

    // Publish tracking gate tile (C layout: row = wave*16 + khalf*8 + r, col).
#pragma unroll
    for (int r = 0; r < 8; ++r) {
      int bb = wave * 16 + khalf * 8 + r;
      W.Gbuf[(size_t)bb * 256 + col] = accG[r];
    }
    grid_barrier(barCount, barGen, NWG);

    // ---- Stage B: tracking LSTM elementwise (torch gate order i,f,g,o) ----
    // Biases b_ih + b_hh folded in here instead of the accumulators.
    {
      int gtid = blockIdx.x * NTHREADS + threadIdx.x;
#pragma unroll
      for (int rep = 0; rep < (kB * kTD) / (NWG * NTHREADS); ++rep) {
        int idx = gtid + rep * NWG * NTHREADS;
        int bb = idx >> 6, j = idx & 63;
        const float* Gr = W.Gbuf + (size_t)bb * 256;
        float gi = Gr[j]       + P.b_ih[j]       + P.b_hh[j];
        float gf = Gr[64 + j]  + P.b_ih[64 + j]  + P.b_hh[64 + j];
        float gg = Gr[128 + j] + P.b_ih[128 + j] + P.b_hh[128 + j];
        float go = Gr[192 + j] + P.b_ih[192 + j] + P.b_hh[192 + j];
        float c2 = sigmoidf(gf) * W.tc[idx] + sigmoidf(gi) * tanhf(gg);
        float h2 = sigmoidf(go) * tanhf(c2);
        W.tc[idx]   = c2;
        W.x_bf[idx] = (bf16_t)h2;   // becomes x for this step, th for next step
      }
    }
    grid_barrier(barCount, barGen, NWG);

    // ---- Stage C: tree gates += x @ W_g^T, then tree cell update ----
#pragma unroll
    for (int s = 0; s < 2; ++s) {
      v16bf a = load_a(xrow, s * 32 + k0a);
#pragma unroll
      for (int g = 0; g < 4; ++g)
        accT[g] = wmma_bf16(a, load_b(smem, SLABS_TRK + g * SLABS_TREE + 16 + s, lane), accT[g]);
    }

#pragma unroll
    for (int r = 0; r < 8; ++r) {
      int bb = wave * 16 + khalf * 8 + r;
      int m   = W.msk[t * kB + bb];
      int rs1 = W.sp1[t * kB + bb];
      int rs2 = W.sp2[t * kB + bb];
      int rbp = W.bp [t * kB + bb];
      float ij = sigmoidf(accT[0][r] + bT0);
      float oj = sigmoidf(accT[1][r] + bT1);
      float fj = sigmoidf(accT[2][r] + bT2);
      float uj = tanhf   (accT[3][r] + bT3);
      float cl = W.stk_c[((size_t)bb * kTP1 + rs2) * kD + col];
      float cr = W.stk_c[((size_t)bb * kTP1 + rs1) * kD + col];
      float cj = ij * uj + fj * (cl + cr);
      float hj = oj * tanhf(cj);
      float hb = tokens_h_f32[((size_t)bb * kL + rbp) * kD + col];
      float cb = tokens_c_f32[((size_t)bb * kL + rbp) * kD + col];
      float nh = m ? hj : hb;
      float nc = m ? cj : cb;
      W.stk_h[((size_t)bb * kTP1 + t) * kD + col] = (bf16_t)nh;
      W.stk_c[((size_t)bb * kTP1 + t) * kD + col] = nc;
      out[(size_t)bb * kD + col] = nh;   // final overwrite at t==T is the answer
    }
    grid_barrier(barCount, barGen, NWG);
  }
}

// ---------------------------------------------------------------------------

extern "C" void kernel_launch(void* const* d_in, const int* in_sizes, int n_in,
                              void* d_out, int out_size, void* d_ws, size_t ws_size,
                              hipStream_t stream) {
  (void)in_sizes; (void)n_in; (void)out_size; (void)ws_size;
  // setup_inputs() dict order (params dict expanded in insertion order):
  const float* tokens_h = (const float*)d_in[0];
  const float* tokens_c = (const float*)d_in[1];
  const int*   trans    = (const int*)  d_in[2];
  ParamsPtrs P;
  P.W_i  = (const float*)d_in[3];  P.W_o  = (const float*)d_in[4];  P.W_u = (const float*)d_in[5];
  P.U_il = (const float*)d_in[6];  P.U_ir = (const float*)d_in[7];
  P.U_fl = (const float*)d_in[8];  P.U_fr = (const float*)d_in[9];
  P.U_ol = (const float*)d_in[10]; P.U_or = (const float*)d_in[11];
  P.U_ul = (const float*)d_in[12]; P.U_ur = (const float*)d_in[13];
  P.b_i  = (const float*)d_in[14]; P.b_f  = (const float*)d_in[15];
  P.b_o  = (const float*)d_in[16]; P.b_u  = (const float*)d_in[17];
  P.W_ih = (const float*)d_in[18]; P.W_hh = (const float*)d_in[19];
  P.b_ih = (const float*)d_in[20]; P.b_hh = (const float*)d_in[21];
  const float* th0 = (const float*)d_in[22];
  const float* tc0 = (const float*)d_in[23];

  char* ws = (char*)d_ws;
  auto carve = [&](size_t bytes) {
    char* p = ws;
    ws += (bytes + 255) & ~(size_t)255;
    return p;
  };
  WsPtrs W;
  W.stk_h = (bf16_t*) carve((size_t)kB * kTP1 * kD * 2);
  W.stk_c = (float*)  carve((size_t)kB * kTP1 * kD * 4);
  W.tok_h = (bf16_t*) carve((size_t)kB * kL * kD * 2);
  W.packW = (bf16_t*) carve((size_t)TOTAL_SLABS * 32 * 16 * 2);
  W.x_bf  = (bf16_t*) carve((size_t)kB * kTD * 2);
  W.tc    = (float*)  carve((size_t)kB * kTD * 4);
  W.Gbuf  = (float*)  carve((size_t)kB * 256 * 4);
  W.sp1   = (int*)    carve((size_t)kTP1 * kB * 4);
  W.sp2   = (int*)    carve((size_t)kTP1 * kB * 4);
  W.bp    = (int*)    carve((size_t)kTP1 * kB * 4);
  W.msk   = (int*)    carve((size_t)kTP1 * kB * 4);
  W.ptrq  = (int*)    carve((size_t)kB * kTP1 * 4);
  W.bar   = (unsigned*)carve(256);

  sched_kernel<<<1, kB, 0, stream>>>(trans, W.sp1, W.sp2, W.bp, W.msk, W.ptrq);
  pack_weights<<<80, 256, 0, stream>>>(P, W.packW);
  size_t n = (size_t)kB * kL * kD;
  init_kernel<<<(int)((n + 255) / 256), 256, 0, stream>>>(
      tokens_h, th0, tc0, W.tok_h, W.stk_h, W.stk_c, W.x_bf, W.tc, W.bar);
  spinn_persistent<<<NWG, NTHREADS, LDS_BYTES, stream>>>(
      P, W, tokens_h, tokens_c, (float*)d_out);
}
